// CrossAttention_gl_61083024884308
// MI455X (gfx1250) — matrix operand order repaired and checked by reading, not compiled, not run on hardware
//
#include <hip/hip_runtime.h>
#include <hip/hip_bf16.h>

// ---------------------------------------------------------------------------
// CrossAttention on gfx1250 (wave32). All GEMMs: v_wmma_f32_16x16x32_bf16.
// Data is laid out so every WMMA fragment is a contiguous vector load:
//   A (16x32 bf16): lane L(0-15) = row M=L, K={0..7,16..23}; lane L+16 =
//       row M=L, K={8..15,24..31}  -> two contiguous 16B (bf16) / 32B (f32)
//       chunks per lane.
//   B (32x16 bf16): lane n holds column N=n, K split as for A -> with B stored
//       K-contiguous per column (i.e. transposed weights / v^T), two 16B
//       vector loads per lane per fragment.
//   C/D (16x16 f32): VGPR i = C[row0 + (lane>>4)*8 + i][col0 + (lane&15)].
// Every GEMM wave computes a 16x64 tile (4 accumulators) so one A fragment
// feeds four back-to-back v_wmma.
// ---------------------------------------------------------------------------

typedef __bf16 bf16_t;
typedef __attribute__((ext_vector_type(16))) __bf16 v16bf;
typedef __attribute__((ext_vector_type(8)))  __bf16 v8bf;
typedef __attribute__((ext_vector_type(8)))  float  v8f;

#define WMMA_BF16(a, b, c) \
  __builtin_amdgcn_wmma_f32_16x16x32_bf16(false, (a), false, (b), (short)0, (c), false, false)

// Prefetch into the near (WGP) cache level, not just GL2.
#define PREFETCH_NEAR(p) __builtin_prefetch((p), 0, 3)

// ---- fragment loaders (all vectorized) -------------------------------------

// A fragment from f32 row-major; Arow = &A[row][0] (32B aligned rows)
__device__ __forceinline__ v16bf a_frag_from_f32(const float* __restrict__ Arow,
                                                 int k0, int half) {
  v8f lo = *(const v8f*)(Arow + k0 + half * 8);
  v8f hi = *(const v8f*)(Arow + k0 + 16 + half * 8);
  v16bf a;
#pragma unroll
  for (int j = 0; j < 8; ++j) { a[j] = (bf16_t)lo[j]; a[8 + j] = (bf16_t)hi[j]; }
  return a;
}

// A fragment from bf16 row-major
__device__ __forceinline__ v16bf a_frag_from_bf16(const bf16_t* __restrict__ Arow,
                                                  int k0, int half) {
  v8bf lo = *(const v8bf*)(Arow + k0 + half * 8);
  v8bf hi = *(const v8bf*)(Arow + k0 + 16 + half * 8);
  return __builtin_shufflevector(lo, hi, 0, 1, 2, 3, 4, 5, 6, 7,
                                         8, 9, 10, 11, 12, 13, 14, 15);
}

// B fragment where source is stored column-K-contiguous: Bt[col][kk]
// (transposed weight, or v^T). Two 16-byte vector loads per lane, K={8h..8h+7, 16+8h..16+8h+7}.
__device__ __forceinline__ v16bf b_frag_kcontig(const bf16_t* __restrict__ Bt, int ldk,
                                                int k0, int col, int half) {
  const bf16_t* p = Bt + (size_t)col * ldk + k0 + half * 8;
  v8bf lo = *(const v8bf*)(p);
  v8bf hi = *(const v8bf*)(p + 16);
  return __builtin_shufflevector(lo, hi, 0, 1, 2, 3, 4, 5, 6, 7,
                                         8, 9, 10, 11, 12, 13, 14, 15);
}

__device__ __forceinline__ void store_c_f32(float* __restrict__ C, int ldc,
                                            int row0, int col0, int lane, v8f acc) {
  int n = lane & 15, half = lane >> 4;
  float* Cp = C + (size_t)(row0 + half * 8) * ldc + col0 + n;
#pragma unroll
  for (int i = 0; i < 8; ++i) { *Cp = acc[i]; Cp += ldc; }
}

// ---- K0: weight convert+transpose  W[K][Nc] f32 -> Wt[Nc][K] bf16 ----------

__global__ void wcvt_kernel(const float* __restrict__ W, bf16_t* __restrict__ Wt,
                            int K, int Nc) {
  int idx = blockIdx.x * blockDim.x + threadIdx.x;
  int total = K * Nc;
  if (idx >= total) return;
  int k = idx / Nc, n = idx - k * Nc;
  Wt[(size_t)n * K + k] = (bf16_t)W[idx];
}

// ---- K1: projection  Out = A[M,K] * W[K,Nc] + bias -------------------------
// mode 0: Out bf16 row-major [M][Nc];  mode 1: Out bf16 batch-transposed
// [B][Nc][Nn] (rows are batch-local, M = B*Nn).

__global__ void proj_kernel(const float* __restrict__ A, const bf16_t* __restrict__ Wt,
                            const float* __restrict__ bias, bf16_t* __restrict__ Out,
                            int M, int K, int Nc, int mode, int Nn) {
  int wave = threadIdx.x >> 5, lane = threadIdx.x & 31;
  int ncg = Nc >> 6;                                   // 64-wide column groups
  int t = blockIdx.x * (blockDim.x >> 5) + wave;
  int total = (M >> 4) * ncg;
  if (t >= total) return;                              // wave-uniform guard
  int row0 = (t / ncg) << 4;
  int col0 = (t % ncg) << 6;
  int half = lane >> 4, n = lane & 15;
  const float* Arow = A + (size_t)(row0 + n) * K;
  v8f acc[4] = {{}, {}, {}, {}};
  for (int k0 = 0; k0 < K; k0 += 32) {
    PREFETCH_NEAR(Arow + k0 + 32);
    v16bf a = a_frag_from_f32(Arow, k0, half);
#pragma unroll
    for (int c = 0; c < 4; ++c) {
      v16bf b = b_frag_kcontig(Wt, K, k0, col0 + 16 * c + n, half);
      acc[c] = WMMA_BF16(a, b, acc[c]);
    }
  }
  if (mode == 0) {
#pragma unroll
    for (int c = 0; c < 4; ++c) {
      int col = col0 + 16 * c + n;
      float bs = bias[col];
      bf16_t* Op = Out + (size_t)(row0 + half * 8) * Nc + col;
#pragma unroll
      for (int i = 0; i < 8; ++i) { *Op = (bf16_t)(acc[c][i] + bs); Op += Nc; }
    }
  } else {
    int bb = row0 / Nn, rloc = row0 - bb * Nn;         // batch-local transpose
#pragma unroll
    for (int c = 0; c < 4; ++c) {
      int col = col0 + 16 * c + n;
      float bs = bias[col];
      v8bf pk;
#pragma unroll
      for (int i = 0; i < 8; ++i) pk[i] = (bf16_t)(acc[c][i] + bs);
      *(v8bf*)(Out + ((size_t)bb * Nc + col) * Nn + rloc + half * 8) = pk;
    }
  }
}

// ---- K2: s[b] = q[b] * k[b]^T  (2048x2048, K=64) ---------------------------
// k row-major [Nn][64] is already K-contiguous per output column.

__global__ void scores_kernel(const bf16_t* __restrict__ qg, const bf16_t* __restrict__ kg,
                              float* __restrict__ sout) {
  const int Nn = 2048, F = 64;
  int b = blockIdx.y;
  const bf16_t* q = qg + (size_t)b * Nn * F;
  const bf16_t* k = kg + (size_t)b * Nn * F;
  float* s = sout + (size_t)b * Nn * Nn;
  int wave = threadIdx.x >> 5, lane = threadIdx.x & 31;
  int t = blockIdx.x * (blockDim.x >> 5) + wave;       // 0..4095
  int ncg = Nn >> 6;                                   // 32 column groups
  int row0 = (t / ncg) << 4;
  int col0 = (t % ncg) << 6;
  int half = lane >> 4, n = lane & 15;
  const bf16_t* qrow = q + (size_t)(row0 + n) * F;
  v8f acc[4] = {{}, {}, {}, {}};
#pragma unroll
  for (int k0 = 0; k0 < F; k0 += 32) {
    v16bf a = a_frag_from_bf16(qrow, k0, half);
#pragma unroll
    for (int c = 0; c < 4; ++c) {
      v16bf bb = b_frag_kcontig(k, F, k0, col0 + 16 * c + n, half);
      acc[c] = WMMA_BF16(a, bb, acc[c]);
    }
  }
#pragma unroll
  for (int c = 0; c < 4; ++c) store_c_f32(s, Nn, row0, col0 + 16 * c, lane, acc[c]);
}

// ---- K3: softmax row statistics (max, sum of exp) --------------------------

__global__ void rowstats_kernel(const float* __restrict__ s,
                                float* __restrict__ rowmax, float* __restrict__ rowsum) {
  __shared__ float red[256];
  const int Nn = 2048;
  int row = blockIdx.x;                                // 0..16383 == b*2048+n
  const float* sr = s + (size_t)row * Nn;
  float m = -3.402823466e38f;
  for (int i = threadIdx.x; i < Nn; i += 256) m = fmaxf(m, sr[i]);
  red[threadIdx.x] = m;
  __syncthreads();
  for (int off = 128; off > 0; off >>= 1) {
    if (threadIdx.x < off) red[threadIdx.x] = fmaxf(red[threadIdx.x], red[threadIdx.x + off]);
    __syncthreads();
  }
  m = red[0];
  __syncthreads();
  float sum = 0.f;
  for (int i = threadIdx.x; i < Nn; i += 256) sum += __expf(sr[i] - m);
  red[threadIdx.x] = sum;
  __syncthreads();
  for (int off = 128; off > 0; off >>= 1) {
    if (threadIdx.x < off) red[threadIdx.x] += red[threadIdx.x + off];
    __syncthreads();
  }
  if (threadIdx.x == 0) { rowmax[row] = m; rowsum[row] = red[0]; }
}

// ---- K4: o[b] = softmax(s[b]) * v[b]^T-layout  (beta formed on the fly) ----
// v is stored transposed [C][Nn] per batch -> B fragments are two 16B vector
// loads. o stored bf16 row-major [M][C] for the final projection.

__global__ void attnv_kernel(const float* __restrict__ sg, const float* __restrict__ rowmaxg,
                             const float* __restrict__ rowsumg, const bf16_t* __restrict__ vtg,
                             bf16_t* __restrict__ og) {
  const int Nn = 2048, C = 512;
  int b = blockIdx.y;
  const float*  s    = sg  + (size_t)b * Nn * Nn;
  const bf16_t* vt   = vtg + (size_t)b * C * Nn;       // [C][Nn]
  bf16_t*       o    = og  + (size_t)b * Nn * C;       // [Nn][C]
  const float*  rmax = rowmaxg + (size_t)b * Nn;
  const float*  rsum = rowsumg + (size_t)b * Nn;
  int wave = threadIdx.x >> 5, lane = threadIdx.x & 31;
  int t = blockIdx.x * (blockDim.x >> 5) + wave;       // 0..1023
  int cg   = t & 7;                                    // 8 column groups of 64
  int row0 = (t >> 3) << 4;                            // 128 row tiles
  int col0 = cg * 64;
  int half = lane >> 4, n = lane & 15;
  int row = row0 + n;
  const float* Sr = s + (size_t)row * Nn;
  float mx  = rmax[row];
  float inv = 1.0f / rsum[row];
  v8f acc[4] = {{}, {}, {}, {}};
  for (int k0 = 0; k0 < Nn; k0 += 32) {
    PREFETCH_NEAR(Sr + k0 + 32);
    v8f lo = *(const v8f*)(Sr + k0 + half * 8);
    v8f hi = *(const v8f*)(Sr + k0 + 16 + half * 8);
    v16bf a;
#pragma unroll
    for (int j = 0; j < 8; ++j) {
      a[j]     = (bf16_t)(__expf(lo[j] - mx) * inv);
      a[8 + j] = (bf16_t)(__expf(hi[j] - mx) * inv);
    }
#pragma unroll
    for (int c = 0; c < 4; ++c) {
      v16bf bb = b_frag_kcontig(vt, Nn, k0, col0 + 16 * c + n, half);
      acc[c] = WMMA_BF16(a, bb, acc[c]);
    }
  }
#pragma unroll
  for (int c = 0; c < 4; ++c) {
    int col = col0 + 16 * c + n;
    bf16_t* Op = o + (size_t)(row0 + half * 8) * C + col;
#pragma unroll
    for (int i = 0; i < 8; ++i) { *Op = (bf16_t)acc[c][i]; Op += C; }
  }
}

// ---- K5: out = gamma * (o * Wo + bias_o) + lam * x  (+ gamma scalar) -------

__global__ void outproj_kernel(const bf16_t* __restrict__ O, const bf16_t* __restrict__ WoT,
                               const float* __restrict__ bias_o, const float* __restrict__ x,
                               const float* __restrict__ gamma, const float* __restrict__ lam,
                               float* __restrict__ out, size_t gamma_idx) {
  const int K = 512, Nc = 512, M = 16384;
  int wave = threadIdx.x >> 5, lane = threadIdx.x & 31;
  int t = blockIdx.x * (blockDim.x >> 5) + wave;       // 0..8191
  int ncg = Nc >> 6;                                   // 8 column groups
  int total = (M >> 4) * ncg;
  if (t >= total) return;
  int row0 = (t / ncg) << 4;
  int col0 = (t % ncg) << 6;
  int half = lane >> 4, n = lane & 15;
  const bf16_t* Arow = O + (size_t)(row0 + n) * K;
  v8f acc[4] = {{}, {}, {}, {}};
  for (int k0 = 0; k0 < K; k0 += 32) {
    PREFETCH_NEAR(Arow + k0 + 32);
    v16bf a = a_frag_from_bf16(Arow, k0, half);
#pragma unroll
    for (int c = 0; c < 4; ++c) {
      v16bf b = b_frag_kcontig(WoT, K, k0, col0 + 16 * c + n, half);
      acc[c] = WMMA_BF16(a, b, acc[c]);
    }
  }
  float g = gamma[0], l = lam[0];
#pragma unroll
  for (int c = 0; c < 4; ++c) {
    int col = col0 + 16 * c + n;
    float bo = bias_o[col];
    size_t off = (size_t)(row0 + half * 8) * Nc + col;
    float* Op = out + off;
    const float* Xp = x + off;
#pragma unroll
    for (int i = 0; i < 8; ++i) {
      *Op = g * (acc[c][i] + bo) + l * (*Xp);
      Op += Nc; Xp += Nc;
    }
  }
  if (blockIdx.x == 0 && threadIdx.x == 0) out[gamma_idx] = g;
}

// ---------------------------------------------------------------------------

extern "C" void kernel_launch(void* const* d_in, const int* in_sizes, int n_in,
                              void* d_out, int out_size, void* d_ws, size_t ws_size,
                              hipStream_t stream) {
  const int B = 8, Nn = 2048, CH = 512, F = 64;
  const size_t OUT_ELEMS = (size_t)B * Nn * CH;        // 8,388,608
  const size_t S_ELEMS   = (size_t)B * Nn * Nn;        // 33,554,432

  const float* h     = (const float*)d_in[0];
  const float* x     = (const float*)d_in[1];
  const float* Wq    = (const float*)d_in[2];
  const float* Wk    = (const float*)d_in[3];
  const float* Wv    = (const float*)d_in[4];
  const float* Wo    = (const float*)d_in[5];
  const float* bq    = (const float*)d_in[6];
  const float* bk    = (const float*)d_in[7];
  const float* bv    = (const float*)d_in[8];
  const float* bo    = (const float*)d_in[9];
  const float* gamma = (const float*)d_in[10];
  const float* lam   = (const float*)d_in[11];

  float* out = (float*)d_out;
  float* s   = out + OUT_ELEMS;

  // workspace layout (~38 MB)
  char* ws = (char*)d_ws;
  const size_t MB = (size_t)1 << 20, KBy = (size_t)1 << 10;
  bf16_t* qb   = (bf16_t*)(ws);                        //  2 MB   [16384][64]
  bf16_t* kb   = (bf16_t*)(ws + 2 * MB);               //  2 MB   [16384][64]
  bf16_t* vtb  = (bf16_t*)(ws + 4 * MB);               // 16 MB   [B][512][2048]
  bf16_t* wqT  = (bf16_t*)(ws + 20 * MB);              // 64 KB   [64][512]
  bf16_t* wkT  = (bf16_t*)(ws + 20 * MB + 64 * KBy);   // 64 KB
  bf16_t* wvT  = (bf16_t*)(ws + 20 * MB + 128 * KBy);  // 512 KB  [512][512]
  bf16_t* woT  = (bf16_t*)(ws + 20 * MB + 640 * KBy);  // 512 KB
  float*  rmax = (float*) (ws + 20 * MB + 1152 * KBy); // 64 KB
  float*  rsum = (float*) (ws + 20 * MB + 1216 * KBy); // 64 KB
  bf16_t* ob   = (bf16_t*)(ws + 22 * MB);              // 16 MB   [16384][512]

  dim3 blk(256);  // 8 waves per block (wave32)

  // K0: weight convert + transpose to bf16 [N][K]
  wcvt_kernel<<<(CH * F + 255) / 256, blk, 0, stream>>>(Wq, wqT, CH, F);
  wcvt_kernel<<<(CH * F + 255) / 256, blk, 0, stream>>>(Wk, wkT, CH, F);
  wcvt_kernel<<<(CH * CH + 255) / 256, blk, 0, stream>>>(Wv, wvT, CH, CH);
  wcvt_kernel<<<(CH * CH + 255) / 256, blk, 0, stream>>>(Wo, woT, CH, CH);

  // K1: projections (wave = 16x64 tile). q/k: 1024 tasks; v: 8192 tasks.
  proj_kernel<<<128,  blk, 0, stream>>>(h, wqT, bq, qb,  B * Nn, CH, F,  0, Nn);
  proj_kernel<<<128,  blk, 0, stream>>>(x, wkT, bk, kb,  B * Nn, CH, F,  0, Nn);
  proj_kernel<<<1024, blk, 0, stream>>>(x, wvT, bv, vtb, B * Nn, CH, CH, 1, Nn);

  // K2: s = q k^T : 4096 wave-tasks per batch / 8 = 512 blocks, y = batch
  scores_kernel<<<dim3(512, B), blk, 0, stream>>>(qb, kb, s);

  // K3: softmax row stats over 2048-wide rows
  rowstats_kernel<<<B * Nn, blk, 0, stream>>>(s, rmax, rsum);

  // K4: o = softmax(s) v : 1024 wave-tasks per batch / 8 = 128 blocks
  attnv_kernel<<<dim3(128, B), blk, 0, stream>>>(s, rmax, rsum, vtb, ob);

  // K5: out = gamma*(o Wo + b) + lam*x ; writes gamma scalar output too
  outproj_kernel<<<1024, blk, 0, stream>>>(ob, woT, bo, x, gamma, lam, out,
                                           OUT_ELEMS + S_ELEMS);
}
